// DGMG_15083925143991
// MI455X (gfx1250) — matrix operand order repaired and checked by reading, not compiled, not run on hardware
//
#include <hip/hip_runtime.h>
#include <hip/hip_bf16.h>
#include <math.h>

// ---------------------------------------------------------------------------
// DGMG AddEdge log-prob, factored form:
//   s[b]    = sum_{n in b} gate_n * hv[n]          (gate_n = sigmoid(hv_n.Wg+bg))
//   gsum[b] = sum_{n in b} gate_n
//   logit_b = (s[b] @ Wp) . We[:256] + gsum[b]*(bp.We[:256])
//             + hv[last_idx[b]] . We[256:384] + be
//   out_b   = log_sigmoid(a_b ? logit : -logit)
// ---------------------------------------------------------------------------

#define DNODE 128          // node hidden size
#define GHID 256           // graph hidden size = 2*DNODE
#define TILE_NODES 64      // nodes per block in the streaming pass
#define PAD 132            // LDS row stride (floats): 16B-aligned float4 stores,
                           // bank = (4n + d) % 64 -> conflict-free in dim-major phase

typedef __attribute__((ext_vector_type(16))) __bf16 v16bf;
typedef __attribute__((ext_vector_type(8)))  float  v8f;

// ---------------------------------------------------------------------------
// Kernel 0: zero the segment accumulators (s and gsum are accumulated with
// atomics, and the harness does not re-zero the workspace between replays).
// ---------------------------------------------------------------------------
__global__ void dgmg_zero_kernel(float* __restrict__ p, int n) {
  int i = blockIdx.x * blockDim.x + threadIdx.x;
  int stride = gridDim.x * blockDim.x;
  for (; i < n; i += stride) p[i] = 0.0f;
}

// ---------------------------------------------------------------------------
// Kernel 1: bandwidth-bound streaming pass over hv (reads each element once).
// Per block: stage 64 nodes x 128 dims into padded LDS (float4, coalesced),
// compute gates node-per-thread, then dim-per-thread run-length segment
// accumulation exploiting sorted seg_ids (atomics only at run boundaries).
// ---------------------------------------------------------------------------
__global__ void dgmg_gate_segsum_kernel(const float* __restrict__ hv,
                                        const float* __restrict__ Wg,
                                        const float* __restrict__ bg,
                                        const int*   __restrict__ seg_ids,
                                        float* __restrict__ s,     // [B, 128]
                                        float* __restrict__ gsum,  // [B]
                                        int N) {
  __shared__ float tile[TILE_NODES * PAD];
  __shared__ float sh_wg[DNODE];
  __shared__ float sh_gate[TILE_NODES];
  __shared__ int   sh_sid[TILE_NODES];

  const int t = threadIdx.x;                 // 0..127
  const int base = blockIdx.x * TILE_NODES;
  const int nNodes = min(TILE_NODES, N - base);
  if (nNodes <= 0) return;

  sh_wg[t] = Wg[t];
  if (t < nNodes) sh_sid[t] = seg_ids[base + t];

  // Stage hv tile: nNodes*32 float4's, fully coalesced.
  const float4* src = (const float4*)(hv + (size_t)base * DNODE);
  const int totalVec = nNodes * (DNODE / 4);
  #pragma unroll
  for (int i = 0; i < TILE_NODES / 4; ++i) {   // 16 iters * 128 threads = 2048
    int vi = i * 128 + t;
    if (vi < totalVec) {
      float4 v = src[vi];
      int n = vi >> 5;                         // 32 float4 per node
      int j = vi & 31;
      float* dst = &tile[n * PAD + j * 4];
      dst[0] = v.x; dst[1] = v.y; dst[2] = v.z; dst[3] = v.w;
    }
  }
  __syncthreads();

  // Gate phase: thread t (< 64) handles node t.
  if (t < TILE_NODES) {
    float gate = 0.0f;
    if (t < nNodes) {
      float dot = 0.0f;
      #pragma unroll 8
      for (int d = 0; d < DNODE; ++d)
        dot = fmaf(tile[t * PAD + d], sh_wg[d], dot);
      gate = 1.0f / (1.0f + __expf(-(dot + bg[0])));
    }
    sh_gate[t] = gate;
  }
  __syncthreads();

  // Accumulation phase: thread t owns dimension d = t. seg_ids are sorted, so
  // keep a running accumulator per run and flush with one atomic per boundary.
  float acc = 0.0f;
  float gacc = 0.0f;
  int cur = sh_sid[0];
  for (int n = 0; n < nNodes; ++n) {
    int sid = sh_sid[n];
    if (sid != cur) {
      atomicAdd(&s[(size_t)cur * DNODE + t], acc);
      if (t == 0) atomicAdd(&gsum[cur], gacc);
      acc = 0.0f; gacc = 0.0f;
      cur = sid;
    }
    float g = sh_gate[n];
    acc = fmaf(g, tile[n * PAD + t], acc);
    gacc += g;
  }
  atomicAdd(&s[(size_t)cur * DNODE + t], acc);
  if (t == 0) atomicAdd(&gsum[cur], gacc);
}

// ---------------------------------------------------------------------------
// Kernel 2: src_dot[b] = hv[last_idx[b]] . We[256:384]
// ---------------------------------------------------------------------------
__global__ void dgmg_srcdot_kernel(const float* __restrict__ hv,
                                   const float* __restrict__ We,
                                   const int*   __restrict__ last_idx,
                                   float* __restrict__ srcdot) {
  const int b = blockIdx.x;
  const int t = threadIdx.x;                 // 128
  const int idx = last_idx[b];
  float v = hv[(size_t)idx * DNODE + t] * We[GHID + t];
  #pragma unroll
  for (int m = 16; m > 0; m >>= 1) v += __shfl_xor(v, m, 32);
  __shared__ float part[4];
  if ((t & 31) == 0) part[t >> 5] = v;
  __syncthreads();
  if (t == 0) srcdot[b] = part[0] + part[1] + part[2] + part[3];
}

// ---------------------------------------------------------------------------
// Kernel 2b: scalar c = bp . We[:256]
// ---------------------------------------------------------------------------
__global__ void dgmg_bpwe_kernel(const float* __restrict__ bp,
                                 const float* __restrict__ We,
                                 float* __restrict__ cbpwe) {
  const int t = threadIdx.x;                 // 256
  float v = bp[t] * We[t];
  #pragma unroll
  for (int m = 16; m > 0; m >>= 1) v += __shfl_xor(v, m, 32);
  __shared__ float part[8];
  if ((t & 31) == 0) part[t >> 5] = v;
  __syncthreads();
  if (t == 0) {
    float sum = 0.0f;
    #pragma unroll
    for (int i = 0; i < 8; ++i) sum += part[i];
    *cbpwe = sum;
  }
}

// ---------------------------------------------------------------------------
// Kernel 3: WMMA GEMM  s[16x128] @ Wp[128x256]  with fused epilogue.
// One wave (32 lanes) per 16-graph stripe. f32 inputs are split into
// bf16 hi/lo and combined as hi*hi + hi*lo + lo*hi (f32 accumulate) for
// near-f32 accuracy on the fast bf16 WMMA pipe.
// ISA wave32 layouts (CDNA5 ISA §7.12.2):
//   A (16x32 bf16):  row M = lane&15,  K = i + 8*(i>>3) + 8*(lane>>4)
//   B (32x16 bf16):  col N = lane&15,  K = i + 8*(i>>3) + 8*(lane>>4)
//   C (16x16 f32):   VGPR r -> M = r + 8*(lane>>4), N = lane&15
// ---------------------------------------------------------------------------
__global__ void dgmg_wmma_logit_kernel(const float* __restrict__ s,      // [B,128]
                                       const float* __restrict__ Wp,     // [128,256]
                                       const float* __restrict__ We,     // [384]
                                       const float* __restrict__ be,     // [1]
                                       const float* __restrict__ gsum,   // [B]
                                       const float* __restrict__ srcdot, // [B]
                                       const float* __restrict__ cbpwe,  // [1]
                                       const int*   __restrict__ act,    // [B]
                                       float* __restrict__ out) {        // [B]
  const int lane = threadIdx.x;              // 32 lanes, EXEC all-ones for WMMA
  const int half = lane >> 4;
  const int lrow = lane & 15;
  const int g0 = blockIdx.x * 16;

  // A fragments: 4 K-steps of the 16x128 stripe of s, split hi/lo bf16.
  v16bf a_hi[4], a_lo[4];
  const float* srow = s + (size_t)(g0 + lrow) * DNODE;
  #pragma unroll
  for (int kk = 0; kk < 4; ++kk) {
    #pragma unroll
    for (int i = 0; i < 16; ++i) {
      int K = kk * 32 + i + 8 * (i >> 3) + 8 * half;
      float x = srow[K];
      __bf16 h = (__bf16)x;
      a_hi[kk][i] = h;
      a_lo[kk][i] = (__bf16)(x - (float)h);
    }
  }

  float acc[8] = {0.f, 0.f, 0.f, 0.f, 0.f, 0.f, 0.f, 0.f};

  #pragma unroll
  for (int ct = 0; ct < 16; ++ct) {          // 16 column tiles -> full G=256
    v8f c = {0.f, 0.f, 0.f, 0.f, 0.f, 0.f, 0.f, 0.f};
    #pragma unroll
    for (int kk = 0; kk < 4; ++kk) {
      v16bf b_hi, b_lo;
      #pragma unroll
      for (int i = 0; i < 16; ++i) {
        int K = kk * 32 + i + 8 * (i >> 3) + 8 * half;
        float x = Wp[(size_t)K * GHID + ct * 16 + lrow];
        __bf16 h = (__bf16)x;
        b_hi[i] = h;
        b_lo[i] = (__bf16)(x - (float)h);
      }
      c = __builtin_amdgcn_wmma_f32_16x16x32_bf16(false, a_hi[kk], false, b_hi,
                                                  (short)0, c, false, false);
      c = __builtin_amdgcn_wmma_f32_16x16x32_bf16(false, a_hi[kk], false, b_lo,
                                                  (short)0, c, false, false);
      c = __builtin_amdgcn_wmma_f32_16x16x32_bf16(false, a_lo[kk], false, b_hi,
                                                  (short)0, c, false, false);
    }
    // Fused epilogue: graph_embed tile . We[:256] (no intermediate to HBM).
    float wcol = We[ct * 16 + lrow];
    #pragma unroll
    for (int r = 0; r < 8; ++r) acc[r] = fmaf(c[r], wcol, acc[r]);
  }

  // Row-reduce over N (16 lanes per half-wave hold the same 8 rows).
  #pragma unroll
  for (int r = 0; r < 8; ++r) {
    #pragma unroll
    for (int m = 1; m < 16; m <<= 1) acc[r] += __shfl_xor(acc[r], m, 16);
  }

  if (lrow == 0) {                           // lanes 0 and 16: 8 graphs each
    const float be0 = be[0];
    const float cb = *cbpwe;
    #pragma unroll
    for (int r = 0; r < 8; ++r) {
      int g = g0 + half * 8 + r;
      float logit = acc[r] + gsum[g] * cb + srcdot[g] + be0;
      float x = act[g] ? logit : -logit;     // pick log_probs[:, a]
      // stable log_sigmoid(x) = min(x,0) - log1p(exp(-|x|))
      out[g] = fminf(x, 0.0f) - log1pf(__expf(-fabsf(x)));
    }
  }
}

// ---------------------------------------------------------------------------
// Host-side launch (all on `stream`, graph-capture safe).
// ---------------------------------------------------------------------------
extern "C" void kernel_launch(void* const* d_in, const int* in_sizes, int n_in,
                              void* d_out, int out_size, void* d_ws, size_t ws_size,
                              hipStream_t stream) {
  const float* hv       = (const float*)d_in[0];
  const float* Wg       = (const float*)d_in[1];
  const float* bg       = (const float*)d_in[2];
  const float* Wp       = (const float*)d_in[3];
  const float* bp       = (const float*)d_in[4];
  const float* We       = (const float*)d_in[5];
  const float* be       = (const float*)d_in[6];
  const int*   seg_ids  = (const int*)d_in[7];
  const int*   last_idx = (const int*)d_in[8];
  const int*   act      = (const int*)d_in[9];

  const int N = in_sizes[7];   // nodes
  const int B = in_sizes[8];   // graphs (8192, multiple of 16)

  float* s      = (float*)d_ws;          // [B,128]
  float* gsum   = s + (size_t)B * DNODE; // [B]
  float* srcdot = gsum + B;              // [B]
  float* cbpwe  = srcdot + B;            // [1]

  (void)n_in; (void)out_size; (void)ws_size;

  // 0) zero accumulators (s and gsum are contiguous)
  dgmg_zero_kernel<<<512, 256, 0, stream>>>(s, B * DNODE + B);

  // 1) streaming gated segment-sum over hv (HBM-bound pass)
  int nb1 = (N + TILE_NODES - 1) / TILE_NODES;
  dgmg_gate_segsum_kernel<<<nb1, 128, 0, stream>>>(hv, Wg, bg, seg_ids, s, gsum, N);

  // 2) src embedding dot + 2b) bp.We scalar
  dgmg_srcdot_kernel<<<B, 128, 0, stream>>>(hv, We, last_idx, srcdot);
  dgmg_bpwe_kernel<<<1, 256, 0, stream>>>(bp, We, cbpwe);

  // 3) WMMA GEMM + fused logit / log-sigmoid epilogue
  dgmg_wmma_logit_kernel<<<B / 16, 32, 0, stream>>>(s, Wp, We, be, gsum, srcdot,
                                                    cbpwe, act, (float*)d_out);
}
